// SingleInputFusionLayer_23467701305776
// MI455X (gfx1250) — matrix operand order repaired and checked by reading, not compiled, not run on hardware
//
#include <hip/hip_runtime.h>
#include <hip/hip_bf16.h>
#include <math.h>

// ---------------------------------------------------------------------------
// MI455X / gfx1250 transformer block: bf16 WMMA everywhere, fp32 accumulate.
// GEMM operands pre-converted to bf16 (b128 fragment loads); V transposed
// per-head; attention bias (mask + adjacency) precomputed once in f32.
// ---------------------------------------------------------------------------

typedef __attribute__((ext_vector_type(16))) __bf16 v16bf;
typedef __attribute__((ext_vector_type(8)))  __bf16 v8bf;
typedef __attribute__((ext_vector_type(8)))  float  v8f;

static constexpr int Bb = 4, Ll = 1024, Dd = 512, Hh = 8, DHh = 64;

__device__ __forceinline__ v16bf concat8(v8bf lo, v8bf hi) {
  return __builtin_shufflevector(lo, hi, 0, 1, 2, 3, 4, 5, 6, 7,
                                 8, 9, 10, 11, 12, 13, 14, 15);
}

// ---- cross-lane helpers (wave32) ------------------------------------------
__device__ __forceinline__ float grp16_max(float v) {
  v = fmaxf(v, __shfl_xor(v, 1));
  v = fmaxf(v, __shfl_xor(v, 2));
  v = fmaxf(v, __shfl_xor(v, 4));
  v = fmaxf(v, __shfl_xor(v, 8));
  return v;
}
__device__ __forceinline__ float grp16_sum(float v) {
  v += __shfl_xor(v, 1);
  v += __shfl_xor(v, 2);
  v += __shfl_xor(v, 4);
  v += __shfl_xor(v, 8);
  return v;
}

// ---------------------------------------------------------------------------
// f32 -> bf16 elementwise conversion (4-wide).
// ---------------------------------------------------------------------------
__global__ __launch_bounds__(256) void cvt_f32_bf16(
    const float* __restrict__ in, __bf16* __restrict__ out, int n4) {
  const int i = blockIdx.x * blockDim.x + threadIdx.x;
  if (i >= n4) return;
  const float4 v = ((const float4*)in)[i];
  __bf16* o = out + (size_t)i * 4;
  o[0] = (__bf16)v.x;
  o[1] = (__bf16)v.y;
  o[2] = (__bf16)v.z;
  o[3] = (__bf16)v.w;
}

// ---------------------------------------------------------------------------
// bf16-WMMA GEMM:  out[M,N] = act(A[M,K] * W[N,K]^T + bias) (+resid)
// Wave tile 32x64 (2x4 of 16x16 wmma accumulators), K-step 32.
// ---------------------------------------------------------------------------
template <bool OUT_BF16, bool GELU, bool RESID>
__global__ __launch_bounds__(256) void gemm_bf16_wmma(
    const __bf16* __restrict__ A, const __bf16* __restrict__ W,
    const float* __restrict__ bias, const float* __restrict__ resid,
    void* __restrict__ outv, int M, int N, int K) {
  const int lane = threadIdx.x & 31;
  const int wid  = (blockIdx.x * blockDim.x + threadIdx.x) >> 5;
  const int tilesN = N >> 6;
  const int tilesM = M >> 5;
  if (wid >= tilesM * tilesN) return;
  const int tm = (wid / tilesN) << 5;
  const int tn = (wid % tilesN) << 6;

  const int mloc = lane & 15;
  const int aklo = (lane >> 4) << 4;   // A-frag K offset per half-wave: 0 / 16
  const int bklo = (lane >> 4) << 4;   // B-frag K offset per half-wave: 0 / 16

  const __bf16* aptr[2];
#pragma unroll
  for (int mi = 0; mi < 2; ++mi)
    aptr[mi] = A + (size_t)(tm + mi * 16 + mloc) * K + aklo;
  const __bf16* bptr[4];
#pragma unroll
  for (int ni = 0; ni < 4; ++ni)
    bptr[ni] = W + (size_t)(tn + ni * 16 + mloc) * K + bklo;

  v8f acc[2][4] = {};

  for (int k0 = 0; k0 < K; k0 += 32) {
    v16bf af[2];
#pragma unroll
    for (int mi = 0; mi < 2; ++mi)
      af[mi] = concat8(*(const v8bf*)(aptr[mi] + k0),
                       *(const v8bf*)(aptr[mi] + k0 + 8));
    v16bf bfm[4];
#pragma unroll
    for (int ni = 0; ni < 4; ++ni)
      bfm[ni] = *(const v16bf*)(bptr[ni] + k0);

    // speculative prefetch of the K-tile two iterations ahead
    __builtin_prefetch(aptr[0] + k0 + 64, 0, 1);
    __builtin_prefetch(bptr[0] + k0 + 64, 0, 1);

#pragma unroll
    for (int mi = 0; mi < 2; ++mi)
#pragma unroll
      for (int ni = 0; ni < 4; ++ni)
        acc[mi][ni] = __builtin_amdgcn_wmma_f32_16x16x32_bf16(
            false, af[mi], false, bfm[ni], (short)0, acc[mi][ni], false, false);
  }

  const int rowhi = (lane >> 4) << 3;  // C-layout: upper half-wave holds M+8
#pragma unroll
  for (int mi = 0; mi < 2; ++mi) {
#pragma unroll
    for (int ni = 0; ni < 4; ++ni) {
      const int n = tn + ni * 16 + (lane & 15);
      const float bv = bias ? bias[n] : 0.0f;
#pragma unroll
      for (int r = 0; r < 8; ++r) {
        const int m = tm + mi * 16 + rowhi + r;
        float v = acc[mi][ni][r] + bv;
        if (GELU) v = 0.5f * v * (1.0f + erff(v * 0.70710678118f));
        if (RESID) v += resid[(size_t)m * N + n];
        if (OUT_BF16) ((__bf16*)outv)[(size_t)m * N + n] = (__bf16)v;
        else          ((float*)outv)[(size_t)m * N + n]  = v;
      }
    }
  }
}

// ---------------------------------------------------------------------------
// LayerNorm: one wave per row of D=512, bf16 output for downstream WMMA.
// ---------------------------------------------------------------------------
__global__ __launch_bounds__(256) void layernorm_bf16(
    const float* __restrict__ x, const float* __restrict__ g,
    const float* __restrict__ b, __bf16* __restrict__ out, int rows, int D) {
  const int lane = threadIdx.x & 31;
  const int row  = (blockIdx.x * blockDim.x + threadIdx.x) >> 5;
  if (row >= rows) return;
  const float4* xr4 = (const float4*)(x + (size_t)row * D);
  const int D4 = D >> 2;
  float s = 0.f, sq = 0.f;
  for (int c = lane; c < D4; c += 32) {
    const float4 v = xr4[c];
    s  += v.x + v.y + v.z + v.w;
    sq += v.x * v.x + v.y * v.y + v.z * v.z + v.w * v.w;
  }
#pragma unroll
  for (int m = 1; m < 32; m <<= 1) {
    s  += __shfl_xor(s, m);
    sq += __shfl_xor(sq, m);
  }
  const float mu   = s / D;
  const float var  = sq / D - mu * mu;
  const float rstd = rsqrtf(var + 1e-5f);
  __bf16* orow = out + (size_t)row * D;
  for (int c = lane; c < D4; c += 32) {
    const float4 v = xr4[c];
    const int o = c * 4;
    orow[o + 0] = (__bf16)((v.x - mu) * rstd * g[o + 0] + b[o + 0]);
    orow[o + 1] = (__bf16)((v.y - mu) * rstd * g[o + 1] + b[o + 1]);
    orow[o + 2] = (__bf16)((v.z - mu) * rstd * g[o + 2] + b[o + 2]);
    orow[o + 3] = (__bf16)((v.w - mu) * rstd * g[o + 3] + b[o + 3]);
  }
}

// ---------------------------------------------------------------------------
// row_all: AND-reduce each mask row (for ensure_self_loops_in_mask).
// ---------------------------------------------------------------------------
__global__ __launch_bounds__(256) void rowall_kernel(
    const unsigned char* __restrict__ mask, unsigned char* __restrict__ rowall,
    int rows, int L) {
  const int lane = threadIdx.x & 31;
  const int row  = (blockIdx.x * blockDim.x + threadIdx.x) >> 5;
  if (row >= rows) return;
  const unsigned char* mr = mask + (size_t)row * L;
  int all = 1;
  for (int c = lane; c < L; c += 32) all &= (mr[c] != 0);
#pragma unroll
  for (int m = 1; m < 32; m <<= 1) all &= __shfl_xor(all, m);
  if (lane == 0) rowall[row] = (unsigned char)all;
}

// ---------------------------------------------------------------------------
// Precompute attention bias (shared by all 8 heads, resident in L2):
//   bias[b,q,k] = (mask && !(q==k && row_all)) * -1e4 + e^ls * clip(adj,0,1)
// ---------------------------------------------------------------------------
__global__ __launch_bounds__(256) void bias_kernel(
    const unsigned char* __restrict__ mask, const float* __restrict__ adj,
    const unsigned char* __restrict__ rowall,
    const float* __restrict__ log_scale, float* __restrict__ bias) {
  const int i4 = blockIdx.x * blockDim.x + threadIdx.x;
  const size_t total = (size_t)Bb * Ll * Ll;
  const size_t e = (size_t)i4 * 4;
  if (e >= total) return;
  const float scale = __expf(log_scale[0]);
  const size_t row = e >> 10;       // b*L + q   (L == 1024)
  const int q  = (int)(row & (Ll - 1));
  const int k0 = (int)(e & (Ll - 1));
  const int ra = rowall[row];
  const uchar4 mv = *(const uchar4*)(mask + e);
  const float4 av = *(const float4*)(adj + e);
  float4 o;
  o.x = (((mv.x != 0) && !((q == k0 + 0) && ra)) ? -10000.f : 0.f) +
        scale * fminf(fmaxf(av.x, 0.f), 1.f);
  o.y = (((mv.y != 0) && !((q == k0 + 1) && ra)) ? -10000.f : 0.f) +
        scale * fminf(fmaxf(av.y, 0.f), 1.f);
  o.z = (((mv.z != 0) && !((q == k0 + 2) && ra)) ? -10000.f : 0.f) +
        scale * fminf(fmaxf(av.z, 0.f), 1.f);
  o.w = (((mv.w != 0) && !((q == k0 + 3) && ra)) ? -10000.f : 0.f) +
        scale * fminf(fmaxf(av.w, 0.f), 1.f);
  *(float4*)(bias + e) = o;
}

// ---------------------------------------------------------------------------
// Per-head V transpose: qkv V-section [B*L, D] -> vt [B,H,DH,L] (bf16).
// Tiled 64x64 LDS transpose; reads and writes fully coalesced.
// ---------------------------------------------------------------------------
__global__ __launch_bounds__(256) void vtrans_kernel(
    const __bf16* __restrict__ qkv, __bf16* __restrict__ vt) {
  __shared__ __bf16 tile[64][65];
  const int tid = threadIdx.x;
  const int blk = blockIdx.x;     // (b*H + hh)*16 + key-tile
  const int kt  = blk & 15;
  const int bh  = blk >> 4;
  const int b   = bh >> 3;
  const int hh  = bh & 7;
  const size_t rowstride = 3 * Dd;
  const size_t inbase =
      ((size_t)b * Ll + kt * 64) * rowstride + 2 * Dd + hh * DHh;
#pragma unroll
  for (int i = 0; i < 16; ++i) {
    const int idx = i * 256 + tid;
    const int kl = idx >> 6, dh = idx & 63;
    tile[kl][dh] = qkv[inbase + (size_t)kl * rowstride + dh];
  }
  __syncthreads();
  const size_t outbase = (size_t)bh * DHh * Ll + kt * 64;
#pragma unroll
  for (int i = 0; i < 16; ++i) {
    const int idx = i * 256 + tid;
    const int dh = idx >> 6, kl = idx & 63;
    vt[outbase + (size_t)dh * Ll + kl] = tile[kl][dh];
  }
}

// ---------------------------------------------------------------------------
// Flash-style attention: one wave owns 16 queries x full DH=64, streams 32
// keys per iteration. QK^T and P*V via v_wmma_f32_16x16x32_bf16; online
// softmax with shfl row reductions; P transposed C->A layout through LDS.
// Bias read as a single precomputed f32 stream; V from transposed buffer.
// ---------------------------------------------------------------------------
__global__ __launch_bounds__(256) void attn_kernel(
    const __bf16* __restrict__ qkv,       // [B*L, 3*D] (q|k|v)
    const __bf16* __restrict__ vt,        // [B,H,DH,L]
    const float* __restrict__ bias,       // [B, L, L] precomputed
    __bf16* __restrict__ ctxout) {        // [B*L, D]
  __shared__ __align__(16) __bf16 plds[8][16][32];  // 1 KB per wave P staging
  const int lane = threadIdx.x & 31;
  const int w    = threadIdx.x >> 5;
  const int wid  = blockIdx.x * 8 + w;
  const int total = Bb * Hh * (Ll / 16);
  if (wid >= total) return;
  const int b   = wid / (Hh * (Ll / 16));
  const int rem = wid % (Hh * (Ll / 16));
  const int hh  = rem / (Ll / 16);
  const int q0  = (rem % (Ll / 16)) * 16;

  const int mloc = lane & 15;
  const int hi8  = (lane >> 4) << 3;  // 0 / 8
  const int aklo = hi8 << 1;          // A-frag K low offset
  const int bklo = hi8 << 1;          // B-frag K low offset

  const size_t rowstride = 3 * Dd;    // 1536
  const size_t base = (size_t)b * Ll;
  const __bf16* vtb = vt + (size_t)(b * Hh + hh) * DHh * Ll;

  // Q fragments: DH=64 -> two K=32 A-fragments, resident for whole kernel
  v16bf qf[2];
  {
    const __bf16* qp = qkv + (base + q0 + mloc) * rowstride + hh * DHh;
#pragma unroll
    for (int f = 0; f < 2; ++f) {
      const __bf16* p = qp + f * 32 + aklo;
      qf[f] = concat8(*(const v8bf*)p, *(const v8bf*)(p + 8));
    }
  }

  v8f ctx[4] = {};
  float mrun[8], lrun[8];
#pragma unroll
  for (int r = 0; r < 8; ++r) { mrun[r] = -3.0e38f; lrun[r] = 0.f; }

  // per-lane bias pointer: row (q0+hi8), column (k) walks with kb
  const float* bptr = bias + ((size_t)b * Ll + q0 + hi8) * Ll + mloc;

  for (int kb = 0; kb < Ll; kb += 32) {
    // ---- scores: two 16-key tiles, each 2 wmma over DH ----
    v8f sc[2] = {};
#pragma unroll
    for (int t = 0; t < 2; ++t) {
      const int key = kb + t * 16 + mloc;
      const __bf16* kp = qkv + (base + key) * rowstride + Dd + hh * DHh + bklo;
#pragma unroll
      for (int f = 0; f < 2; ++f) {
        const v16bf kf = *(const v16bf*)(kp + f * 32);
        sc[t] = __builtin_amdgcn_wmma_f32_16x16x32_bf16(
            false, qf[f], false, kf, (short)0, sc[t], false, false);
      }
    }
    // ---- bias + online softmax (C-layout: lane=key, vgpr=query row) ----
    float p0[8], p1[8], alpha[8];
#pragma unroll
    for (int r = 0; r < 8; ++r) {
      const float* bp = bptr + (size_t)r * Ll + kb;
      const float s0 = sc[0][r] * 0.125f + bp[0];    // 1/sqrt(64)
      const float s1 = sc[1][r] * 0.125f + bp[16];
      const float tmax = grp16_max(fmaxf(s0, s1));
      const float nm   = fmaxf(mrun[r], tmax);
      alpha[r] = __expf(mrun[r] - nm);
      mrun[r]  = nm;
      p0[r] = __expf(s0 - nm);
      p1[r] = __expf(s1 - nm);
      lrun[r] = lrun[r] * alpha[r] + grp16_sum(p0[r] + p1[r]);
    }
#pragma unroll
    for (int j = 0; j < 4; ++j)
#pragma unroll
      for (int r = 0; r < 8; ++r) ctx[j][r] *= alpha[r];

    // ---- transpose P: C-layout -> LDS row-major -> A-fragment ----
#pragma unroll
    for (int r = 0; r < 8; ++r) {
      plds[w][hi8 + r][mloc]      = (__bf16)p0[r];
      plds[w][hi8 + r][16 + mloc] = (__bf16)p1[r];
    }
    const v16bf pf = concat8(*(const v8bf*)&plds[w][mloc][aklo],
                             *(const v8bf*)&plds[w][mloc][8 + aklo]);
    // ---- ctx += P * V^T (4 output n-tiles of 16 dh, contiguous loads) ----
#pragma unroll
    for (int j = 0; j < 4; ++j) {
      const int dh = j * 16 + mloc;
      const __bf16* vp = vtb + (size_t)dh * Ll + kb + bklo;
      const v16bf vf = *(const v16bf*)vp;
      __builtin_prefetch(vp + 32, 0, 1);  // next key block
      ctx[j] = __builtin_amdgcn_wmma_f32_16x16x32_bf16(
          false, pf, false, vf, (short)0, ctx[j], false, false);
    }
  }

  // ---- finalize: divide by softmax sum, store bf16 ctx [B*L, D] ----
#pragma unroll
  for (int r = 0; r < 8; ++r) {
    const int qm = q0 + hi8 + r;
    const float inv = 1.0f / lrun[r];
#pragma unroll
    for (int j = 0; j < 4; ++j)
      ctxout[(base + qm) * Dd + hh * DHh + j * 16 + mloc] =
          (__bf16)(ctx[j][r] * inv);
  }
}

// ---------------------------------------------------------------------------
// Launch. Scratch layout (~85 MB):
//   x0 f32 8MB | x0n bf16 4MB | qkv bf16 12MB | ctx bf16 4MB |
//   x1 f32 8MB | x1n bf16 4MB | h bf16 16MB   | rowall 4KB  |
//   xbf 2MB | wfuse 256KB | winproj 1.5MB | woutw 512KB | w1 2MB | w2 2MB |
//   vt 4MB | bias f32 16MB
// ---------------------------------------------------------------------------
extern "C" void kernel_launch(void* const* d_in, const int* in_sizes, int n_in,
                              void* d_out, int out_size, void* d_ws,
                              size_t ws_size, hipStream_t stream) {
  (void)in_sizes; (void)n_in; (void)out_size; (void)ws_size;
  const float*         x      = (const float*)d_in[0];
  const unsigned char* mask   = (const unsigned char*)d_in[1];
  const float*         adj    = (const float*)d_in[2];
  const float*         W_fuse = (const float*)d_in[3];
  const float*         b_fuse = (const float*)d_in[4];
  const float*         in_w   = (const float*)d_in[5];
  const float*         in_b   = (const float*)d_in[6];
  const float*         out_w  = (const float*)d_in[7];
  const float*         out_b  = (const float*)d_in[8];
  const float*         ln1_g  = (const float*)d_in[9];
  const float*         ln1_b  = (const float*)d_in[10];
  const float*         ln2_g  = (const float*)d_in[11];
  const float*         ln2_b  = (const float*)d_in[12];
  const float*         W1     = (const float*)d_in[13];
  const float*         b1     = (const float*)d_in[14];
  const float*         W2     = (const float*)d_in[15];
  const float*         b2     = (const float*)d_in[16];
  const float*         lscale = (const float*)d_in[17];

  char* ws = (char*)d_ws;
  float*         x0   = (float*)(ws);
  __bf16*        x0n  = (__bf16*)(ws + 8388608);
  __bf16*        qkv  = (__bf16*)(ws + 12582912);
  __bf16*        ctxb = (__bf16*)(ws + 25165824);
  float*         x1   = (float*)(ws + 29360128);
  __bf16*        x1n  = (__bf16*)(ws + 37748736);
  __bf16*        hbuf = (__bf16*)(ws + 41943040);
  unsigned char* rall = (unsigned char*)(ws + 58720256);
  __bf16*        xbf  = (__bf16*)(ws + 58724352);  // 4096*256
  __bf16*        wfu  = (__bf16*)(ws + 60821504);  // 512*256
  __bf16*        wip  = (__bf16*)(ws + 61083648);  // 1536*512
  __bf16*        wop  = (__bf16*)(ws + 62656512);  // 512*512
  __bf16*        w1b  = (__bf16*)(ws + 63180800);  // 2048*512
  __bf16*        w2b  = (__bf16*)(ws + 65277952);  // 512*2048
  __bf16*        vt   = (__bf16*)(ws + 67375104);  // 4*8*64*1024
  float*         biasb= (float*)(ws + 71569408);   // 4*1024*1024 f32

  const int M = Bb * Ll;  // 4096
  dim3 blk(256);
  auto cvt = [&](const float* src, __bf16* dst, int n) {
    cvt_f32_bf16<<<dim3((n / 4 + 255) / 256), blk, 0, stream>>>(src, dst, n / 4);
  };

  // 0) one-time bf16 conversions (activations + all weights)
  cvt(x, xbf, M * 256);
  cvt(W_fuse, wfu, 512 * 256);
  cvt(in_w, wip, 1536 * 512);
  cvt(out_w, wop, 512 * 512);
  cvt(W1, w1b, 2048 * 512);
  cvt(W2, w2b, 512 * 2048);

  // 0b) mask row-reduction, then the shared attention bias matrix
  rowall_kernel<<<dim3(512), blk, 0, stream>>>(mask, rall, M, Ll);
  bias_kernel<<<dim3((Bb * Ll * Ll / 4) / 256), blk, 0, stream>>>(
      mask, adj, rall, lscale, biasb);

  // 1) fuse projection: xbf[4096,256] * W_fuse^T + b_fuse -> x0 f32
  gemm_bf16_wmma<false, false, false>
      <<<dim3((128 * 8) / 8), blk, 0, stream>>>(xbf, wfu, b_fuse, nullptr,
                                                x0, M, 512, 256);
  // 2) LN1 -> bf16
  layernorm_bf16<<<dim3(512), blk, 0, stream>>>(x0, ln1_g, ln1_b, x0n, M, 512);
  // 3) QKV -> qkv bf16
  gemm_bf16_wmma<true, false, false>
      <<<dim3((128 * 24) / 8), blk, 0, stream>>>(x0n, wip, in_b, nullptr,
                                                 qkv, M, 1536, 512);
  // 4) per-head V transpose
  vtrans_kernel<<<dim3(512), blk, 0, stream>>>(qkv, vt);
  // 5) attention (flash-style, wmma QK^T + PV)
  attn_kernel<<<dim3(256), blk, 0, stream>>>(qkv, vt, biasb, ctxb);
  // 6) out projection + residual -> x1 f32
  gemm_bf16_wmma<false, false, true>
      <<<dim3((128 * 8) / 8), blk, 0, stream>>>(ctxb, wop, out_b, x0, x1,
                                                M, 512, 512);
  // 7) LN2 -> bf16
  layernorm_bf16<<<dim3(512), blk, 0, stream>>>(x1, ln2_g, ln2_b, x1n, M, 512);
  // 8) FFN1 + exact GELU -> h bf16
  gemm_bf16_wmma<true, true, false>
      <<<dim3((128 * 32) / 8), blk, 0, stream>>>(x1n, w1b, b1, nullptr, hbuf,
                                                 M, 2048, 512);
  // 9) FFN2 + residual -> d_out f32
  gemm_bf16_wmma<false, false, true>
      <<<dim3((128 * 8) / 8), blk, 0, stream>>>(hbuf, w2b, b2, x1,
                                                (float*)d_out, M, 512, 2048);
}
